// SAGENet_14602888806923
// MI455X (gfx1250) — matrix operand order repaired and checked by reading, not compiled, not run on hardware
//
#include <hip/hip_runtime.h>

// ---------------------------------------------------------------------------
// GraphSAGE (mean) x2 + residual sum, tuned for MI455X (gfx1250, wave32, WMMA)
//   t1  = h @ Ws0^T + b0 + mean_nb(h) @ Wn0^T
//   t2  = t1 @ Ws1^T + b1 + mean_nb(t1) @ Wn1^T
//   out = h + t1 + 0.5 * t2
// Aggregation: f32 atomics into L2-resident segment sums (51 MB table << 192MB L2).
// GEMMs: v_wmma_f32_16x16x32_f16, f32 accumulation, operands staged in LDS as f16.
// 64 rows/block (4 row-tiles per wave) -> 4x less weight restaging, 4 independent
// WMMA accumulator chains, B-fragment reuse across row-tiles.
// ---------------------------------------------------------------------------

typedef __attribute__((ext_vector_type(16))) _Float16 v16h;
typedef __attribute__((ext_vector_type(8)))  _Float16 v8h;
typedef __attribute__((ext_vector_type(8)))  float    v8f;

#define DF    128         // feature dim
#define LDW   136         // padded LDS row stride (halfs) -> 272B, avoids bank conflicts
#define ROWS  64          // node rows per block (4 row-tiles of 16)

// ---- fragment loaders (wave32 WMMA 16x16x32 f16 layouts, per CDNA5 ISA §7.12.2) ----

// A (16x32, MxK): lanes 0-15 row M=lane, K = kb+{0..7, 16..23}; lanes 16-31 same row,
// K = kb+{8..15, 24..31}.
__device__ __forceinline__ v16h load_a_frag(const _Float16* row, int kb, int lane) {
  const int koff = (lane >> 4) << 3;               // 0 or 8
  const _Float16* p0 = row + kb + koff;
  v8h lo = *(const v8h*)(p0);
  v8h hi = *(const v8h*)(p0 + 16);
  return __builtin_shufflevector(lo, hi, 0,1,2,3,4,5,6,7,8,9,10,11,12,13,14,15);
}

// B (32x16, KxN): lanes 0-15 col N=lane hold K = kb+{0..7, 16..23}; lanes 16-31 hold K = kb+{8..15, 24..31}.
__device__ __forceinline__ v16h load_b_frag(const _Float16* row, int kb, int lane) {
  const int ks = (lane >> 4) << 3;                 // 0 or 8
  const _Float16* p = row + kb + ks;
  v8h lo = *(const v8h*)(p);
  v8h hi = *(const v8h*)(p + 16);
  return __builtin_shufflevector(lo, hi, 0,1,2,3,4,5,6,7,8,9,10,11,12,13,14,15);
}

// ---------------------------------------------------------------------------
__global__ void zero_f32(float* __restrict__ p, long n) {
  long i = (long)blockIdx.x * blockDim.x + threadIdx.x;
  long stride = (long)gridDim.x * blockDim.x;
  for (; i < n; i += stride) p[i] = 0.0f;
}

// One wave32 per edge: lane l handles features [4l, 4l+4). float4 gather of the
// source row (L2-resident), 4x global_atomic_add_f32 into the destination row.
__global__ __launch_bounds__(256)
void sage_scatter(const float* __restrict__ x, const int* __restrict__ src,
                  const int* __restrict__ dst, float* __restrict__ ssum,
                  float* __restrict__ deg, int E) {
  int e = blockIdx.x * 8 + (threadIdx.x >> 5);
  if (e >= E) return;
  int lane = threadIdx.x & 31;
  int s = src[e];
  int d = dst[e];
  const float4 v = *(const float4*)(x + (size_t)s * DF + lane * 4);
  float* o = ssum + (size_t)d * DF + lane * 4;
  atomicAdd(o + 0, v.x);
  atomicAdd(o + 1, v.y);
  atomicAdd(o + 2, v.z);
  atomicAdd(o + 3, v.w);
  if (deg != nullptr && lane == 0) atomicAdd(deg + d, 1.0f);
}

// Fused SAGE layer: out = X @ Wself^T + bias + (ssum/deg) @ Wneigh^T
// layer2 != 0: out = res0 + X + 0.5 * (that)   [final residual sum, X == t1]
// Block: 256 threads (8 waves), 64 node rows; wave w owns output cols [16w,16w+16)
// and iterates 4 row-tiles with independent accumulators.
__global__ __launch_bounds__(256)
void sage_gemm(const float* __restrict__ X, const float* __restrict__ ssum,
               const float* __restrict__ deg,
               const float* __restrict__ Wself, const float* __restrict__ bias,
               const float* __restrict__ Wneigh,
               float* __restrict__ out, const float* __restrict__ res0,
               int N, int layer2) {
  __shared__ __align__(16) _Float16 sW[DF * LDW];    // 34816 B (one weight matrix)
  __shared__ __align__(16) _Float16 sA[ROWS * LDW];  // 17408 B (one activation tile)

  const int tid   = threadIdx.x;
  const int mbase = blockIdx.x * ROWS;

  // ---- pass 1: self features x W_self ----
  for (int i = tid; i < ROWS * DF; i += 256) {
    int r = i >> 7, c = i & (DF - 1);
    int row = mbase + r;
    float xv = (row < N) ? X[row * DF + c] : 0.0f;
    sA[r * LDW + c] = (_Float16)xv;
  }
  for (int i = tid; i < DF * DF; i += 256) {
    int r = i >> 7, c = i & (DF - 1);
    sW[r * LDW + c] = (_Float16)Wself[i];
  }
  __syncthreads();

  const int lane  = tid & 31;
  const int nbase = (tid >> 5) << 4;
  const int hr    = lane & 15;
  const _Float16* bRow = &sW[(nbase + hr) * LDW];

  v8f acc[4] = {};

#pragma unroll
  for (int kb = 0; kb < DF; kb += 32) {
    v16h b = load_b_frag(bRow, kb, lane);            // reused by 4 row-tiles
#pragma unroll
    for (int rt = 0; rt < 4; ++rt) {
      v16h a = load_a_frag(&sA[(rt * 16 + hr) * LDW], kb, lane);
      acc[rt] = __builtin_amdgcn_wmma_f32_16x16x32_f16(false, a, false, b,
                                                       (short)0, acc[rt],
                                                       false, false);
    }
  }

  // ---- pass 2: neighbor means x W_neigh (restage both LDS buffers) ----
  __syncthreads();
  for (int i = tid; i < ROWS * DF; i += 256) {
    int r = i >> 7, c = i & (DF - 1);
    int row = mbase + r;
    float nv = 0.0f;
    if (row < N) {
      float dg = deg[row];
      nv = (dg > 0.0f) ? ssum[row * DF + c] / dg : 0.0f;
    }
    sA[r * LDW + c] = (_Float16)nv;
  }
  for (int i = tid; i < DF * DF; i += 256) {
    int r = i >> 7, c = i & (DF - 1);
    sW[r * LDW + c] = (_Float16)Wneigh[i];
  }
  __syncthreads();

#pragma unroll
  for (int kb = 0; kb < DF; kb += 32) {
    v16h b = load_b_frag(bRow, kb, lane);
#pragma unroll
    for (int rt = 0; rt < 4; ++rt) {
      v16h a = load_a_frag(&sA[(rt * 16 + hr) * LDW], kb, lane);
      acc[rt] = __builtin_amdgcn_wmma_f32_16x16x32_f16(false, a, false, b,
                                                       (short)0, acc[rt],
                                                       false, false);
    }
  }

  // C/D layout: VGPR r -> M = r (lanes 0-15) or r+8 (lanes 16-31); N = lane&15.
  const int col    = nbase + hr;
  const int rowAdd = (lane >> 4) << 3;
  const float bv   = bias[col];
#pragma unroll
  for (int rt = 0; rt < 4; ++rt) {
#pragma unroll
    for (int r = 0; r < 8; ++r) {
      int row = mbase + rt * 16 + r + rowAdd;
      if (row < N) {
        int g = row * DF + col;
        float v = acc[rt][r] + bv;
        out[g] = layer2 ? (res0[g] + X[g] + 0.5f * v) : v;
      }
    }
  }
}

// ---------------------------------------------------------------------------
extern "C" void kernel_launch(void* const* d_in, const int* in_sizes, int n_in,
                              void* d_out, int out_size, void* d_ws, size_t ws_size,
                              hipStream_t stream) {
  const float* h   = (const float*)d_in[0];
  const int*   src = (const int*)  d_in[1];
  const int*   dst = (const int*)  d_in[2];
  const float* Ws0 = (const float*)d_in[3];
  const float* b0  = (const float*)d_in[4];
  const float* Wn0 = (const float*)d_in[5];
  const float* Ws1 = (const float*)d_in[6];
  const float* b1  = (const float*)d_in[7];
  const float* Wn1 = (const float*)d_in[8];
  float* out = (float*)d_out;              // holds t1 after layer 1, final after layer 2

  const int N = in_sizes[0] / DF;          // 100000
  const int E = in_sizes[1];               // 1600000

  float* ssum = (float*)d_ws;              // N*DF floats
  float* deg  = ssum + (size_t)N * DF;     // N floats (computed once, reused)

  const int zgrid = 2048, zb = 256;
  const int sgrid = (E + 7) / 8;
  const int ggrid = (N + ROWS - 1) / ROWS;

  // ---- layer 1: t1 -> out ----
  zero_f32<<<zgrid, zb, 0, stream>>>(ssum, (long)N * DF + N);
  sage_scatter<<<sgrid, 256, 0, stream>>>(h, src, dst, ssum, deg, E);
  sage_gemm<<<ggrid, 256, 0, stream>>>(h, ssum, deg, Ws0, b0, Wn0,
                                       out, nullptr, N, 0);

  // ---- layer 2 + residual: out = h + t1 + 0.5*t2 (in place over t1) ----
  zero_f32<<<zgrid, zb, 0, stream>>>(ssum, (long)N * DF);
  sage_scatter<<<sgrid, 256, 0, stream>>>(out, src, dst, ssum, nullptr, E);
  sage_gemm<<<ggrid, 256, 0, stream>>>(out, ssum, deg, Ws1, b1, Wn1,
                                       out, h, N, 1);
}
